// PointNetSetAbstractionPN2_63178968924302
// MI455X (gfx1250) — matrix operand order repaired and checked by reading, not compiled, not run on hardware
//
#include <hip/hip_runtime.h>

// ---------------------------------------------------------------------------
// PointNet++ set abstraction for MI455X (gfx1250, wave32).
//   K1 fps_kernel      : 1 workgroup / segment, sequential FPS in 67-dim space
//   K2 gather_kernel   : new_xyz / new_vel / new_offset
//   K3 knn_kernel      : 16-NN per query (wave per query, register top-16)
//   K4 group_gemm      : gather feats -> v_wmma_f32_16x16x32_f16 -> relu -> max
// ---------------------------------------------------------------------------

typedef __attribute__((ext_vector_type(16))) _Float16 v16h;
typedef __attribute__((ext_vector_type(8)))  float    v8f;

#define C_IN   64
#define C_OUT  128
#define NS     16
#define STRIDE 4

// ---------------------------------------------------------------- K1: FPS ---
__global__ __launch_bounds__(1024)
void fps_kernel(const float* __restrict__ xyz, const float* __restrict__ feat,
                int* __restrict__ sampled, int n, int m) {
  const int seg  = blockIdx.x;
  const int tid  = threadIdx.x;
  const int base = seg * n;
  const int lane = tid & 31;
  const int wave = tid >> 5;

  float dist[8];
#pragma unroll
  for (int k = 0; k < 8; ++k) dist[k] = 1e10f;

  __shared__ float qs[68];
  __shared__ float redD[32];
  __shared__ int   redI[32];
  __shared__ int   lastSh;

  int* samp = sampled + seg * m;
  if (tid == 0) samp[0] = 0;   // pointops convention: first pick = 0
  int last = 0;

  for (int it = 1; it < m; ++it) {
    // broadcast the last selected point's 67-dim coords through LDS
    if (tid < 3)       qs[tid] = xyz[(base + last) * 3 + tid];
    else if (tid < 67) qs[tid] = feat[(base + last) * C_IN + (tid - 3)];
    __syncthreads();

    float bestD = -1.0f; int bestI = 0;
#pragma unroll
    for (int k = 0; k < 8; ++k) {
      const int p = k * 1024 + tid;
      if (p < n) {
        const float* xr = xyz + (base + p) * 3;
        float dx = xr[0] - qs[0], dy = xr[1] - qs[1], dz = xr[2] - qs[2];
        float d = dx * dx + dy * dy + dz * dz;
        const float4* fr = (const float4*)(feat + (size_t)(base + p) * C_IN);
#pragma unroll
        for (int c4 = 0; c4 < 16; ++c4) {
          float4 f = fr[c4];
          float a0 = f.x - qs[3 + 4 * c4 + 0];
          float a1 = f.y - qs[3 + 4 * c4 + 1];
          float a2 = f.z - qs[3 + 4 * c4 + 2];
          float a3 = f.w - qs[3 + 4 * c4 + 3];
          d = fmaf(a0, a0, d); d = fmaf(a1, a1, d);
          d = fmaf(a2, a2, d); d = fmaf(a3, a3, d);
        }
        float nd = fminf(dist[k], d);
        dist[k] = nd;
        if (nd > bestD) { bestD = nd; bestI = p; }   // strict > keeps lowest idx
      }
    }
    // wave32 butterfly argmax (tie -> smaller index)
#pragma unroll
    for (int off = 16; off > 0; off >>= 1) {
      float od = __shfl_xor(bestD, off, 32);
      int   oi = __shfl_xor(bestI, off, 32);
      if (od > bestD || (od == bestD && oi < bestI)) { bestD = od; bestI = oi; }
    }
    if (lane == 0) { redD[wave] = bestD; redI[wave] = bestI; }
    __syncthreads();
    if (tid == 0) {
      float bd = redD[0]; int bi = redI[0];
      for (int w = 1; w < 32; ++w)
        if (redD[w] > bd || (redD[w] == bd && redI[w] < bi)) { bd = redD[w]; bi = redI[w]; }
      samp[it] = bi;
      lastSh   = bi;
    }
    __syncthreads();
    last = lastSh;
  }
}

// ------------------------------------------------------------- K2: gather ---
__global__ void gather_kernel(const float* __restrict__ xyz,
                              const float* __restrict__ vel,
                              const int* __restrict__ sampled,
                              const int* __restrict__ offset,
                              float* __restrict__ out_xyz,
                              float* __restrict__ out_vel,
                              float* __restrict__ out_off,
                              int n, int m, int total, int B) {
  const int q = blockIdx.x * blockDim.x + threadIdx.x;
  if (q < total) {
    const int seg = q / m;
    const int g   = seg * n + sampled[q];
    out_xyz[q * 3 + 0] = xyz[g * 3 + 0];
    out_xyz[q * 3 + 1] = xyz[g * 3 + 1];
    out_xyz[q * 3 + 2] = xyz[g * 3 + 2];
    out_vel[q * 3 + 0] = vel[g * 3 + 0];
    out_vel[q * 3 + 1] = vel[g * 3 + 1];
    out_vel[q * 3 + 2] = vel[g * 3 + 2];
  }
  if (q == 0) {
    int prev = 0, acc = 0;
    for (int b = 0; b < B; ++b) {
      acc += (offset[b] - prev) / STRIDE;
      prev = offset[b];
      out_off[b] = (float)acc;
    }
  }
}

// ---------------------------------------------------------------- K3: kNN ---
__global__ __launch_bounds__(256)
void knn_kernel(const float* __restrict__ xyz, const int* __restrict__ sampled,
                int* __restrict__ knn, int n, int m, int total) {
  const int lane = threadIdx.x & 31;
  const int wave = threadIdx.x >> 5;
  int q = blockIdx.x * 8 + wave;
  if (q >= total) q = total - 1;          // clamp: keep all waves in barriers
  const int seg  = q / m;
  const int base = seg * n;
  const int sidx = base + sampled[q];
  const float qx = xyz[sidx * 3 + 0];
  const float qy = xyz[sidx * 3 + 1];
  const float qz = xyz[sidx * 3 + 2];

  float ad[NS]; int ai[NS];
#pragma unroll
  for (int k = 0; k < NS; ++k) { ad[k] = 3.4e38f; ai[k] = 0; }

  for (int p = lane; p < n; p += 32) {
    const float* xr = xyz + (base + p) * 3;
    float dx = xr[0] - qx, dy = xr[1] - qy, dz = xr[2] - qz;
    float d  = fmaf(dx, dx, fmaf(dy, dy, dz * dz));
    if (d < ad[NS - 1]) {
      ad[NS - 1] = d; ai[NS - 1] = p;
#pragma unroll
      for (int k = NS - 1; k > 0; --k) {
        if (ad[k] < ad[k - 1]) {
          float td = ad[k]; ad[k] = ad[k - 1]; ad[k - 1] = td;
          int   ti = ai[k]; ai[k] = ai[k - 1]; ai[k - 1] = ti;
        }
      }
    }
  }

  // merge 32 sorted lists of 16 via LDS heads + wave-min reductions
  __shared__ float sd[8][32 * NS];
  __shared__ int   si[8][32 * NS];
#pragma unroll
  for (int k = 0; k < NS; ++k) {
    sd[wave][lane * NS + k] = ad[k];
    si[wave][lane * NS + k] = ai[k];
  }
  __syncthreads();

  int head = 0;
  for (int r = 0; r < NS; ++r) {
    float cd = sd[wave][lane * NS + head];
    int   ci = si[wave][lane * NS + head];
    int   cl = lane;
#pragma unroll
    for (int off = 16; off > 0; off >>= 1) {
      float od = __shfl_xor(cd, off, 32);
      int   oi = __shfl_xor(ci, off, 32);
      int   ol = __shfl_xor(cl, off, 32);
      if (od < cd || (od == cd && (oi < ci || (oi == ci && ol < cl)))) {
        cd = od; ci = oi; cl = ol;
      }
    }
    if (lane == 0) knn[q * NS + r] = ci;
    if (lane == cl) head++;
  }
}

// ------------------------------------------ K4: gather + WMMA GEMM + pool ---
union AV { v16h v; _Float16 h[16]; };

__global__ __launch_bounds__(256)
void group_gemm_kernel(const float* __restrict__ feat, const float* __restrict__ W,
                       const int* __restrict__ knn, float* __restrict__ out_feat,
                       int n, int m, int total) {
  // W in WMMA-B lane layout, f16: [tile 8][chunk 2][lane 32][elem 16] = 16 KB
  __shared__ __align__(32) _Float16 Bm[8 * 2 * 32 * 16];
  const int tid = threadIdx.x;
  for (int s = tid; s < 512; s += 256) {
    const int t = s >> 6, c = (s >> 5) & 1, l = s & 31;
    const int row = 16 * t + (l & 15);            // output channel (N)
    const int cb  = 32 * c + (l >> 4) * 16;       // K base for this half-wave
    const float4* wr = (const float4*)(W + row * C_IN + cb);
    _Float16* dst = Bm + s * 16;
#pragma unroll
    for (int k4 = 0; k4 < 4; ++k4) {
      float4 f = wr[k4];
      dst[4 * k4 + 0] = (_Float16)f.x; dst[4 * k4 + 1] = (_Float16)f.y;
      dst[4 * k4 + 2] = (_Float16)f.z; dst[4 * k4 + 3] = (_Float16)f.w;
    }
  }
  __syncthreads();

  const int lane = tid & 31;
  const int wave = tid >> 5;
  int q = blockIdx.x * 8 + wave;
  if (q >= total) q = total - 1;                  // duplicate work, same output
  const int seg  = q / m;
  const int base = seg * n;
  const int hi   = lane >> 4;
  const int mrow = lane & 15;                     // neighbor index (M)

  const int nbr = base + knn[q * NS + mrow];
  const float* frow = feat + (size_t)nbr * C_IN;

  // A operand: 16 neighbors x 64 channels, two K=32 chunks, contiguous k = 16*hi + s (same as Bm)
  AV a[2];
#pragma unroll
  for (int c = 0; c < 2; ++c) {
    const float4* f0 = (const float4*)(frow + 32 * c + hi * 16);
    const float4* f1 = (const float4*)(frow + 32 * c + hi * 16 + 8);
    float4 x0 = f0[0], x1 = f0[1], y0 = f1[0], y1 = f1[1];
    a[c].h[0]  = (_Float16)x0.x; a[c].h[1]  = (_Float16)x0.y;
    a[c].h[2]  = (_Float16)x0.z; a[c].h[3]  = (_Float16)x0.w;
    a[c].h[4]  = (_Float16)x1.x; a[c].h[5]  = (_Float16)x1.y;
    a[c].h[6]  = (_Float16)x1.z; a[c].h[7]  = (_Float16)x1.w;
    a[c].h[8]  = (_Float16)y0.x; a[c].h[9]  = (_Float16)y0.y;
    a[c].h[10] = (_Float16)y0.z; a[c].h[11] = (_Float16)y0.w;
    a[c].h[12] = (_Float16)y1.x; a[c].h[13] = (_Float16)y1.y;
    a[c].h[14] = (_Float16)y1.z; a[c].h[15] = (_Float16)y1.w;
  }

  float* orow = out_feat + (size_t)q * C_OUT;
#pragma unroll
  for (int t = 0; t < 8; ++t) {
    v16h b0 = *(const v16h*)(Bm + ((t * 2 + 0) * 32 + lane) * 16);
    v16h b1 = *(const v16h*)(Bm + ((t * 2 + 1) * 32 + lane) * 16);
    v8f acc = {};
    acc = __builtin_amdgcn_wmma_f32_16x16x32_f16(false, a[0].v, false, b0,
                                                 (short)0, acc, false, false);
    acc = __builtin_amdgcn_wmma_f32_16x16x32_f16(false, a[1].v, false, b1,
                                                 (short)0, acc, false, false);
    // relu + max over neighbors: lane holds rows {8*hi..8*hi+7} of column (lane&15)
    float mx = 0.0f;                               // relu floor
#pragma unroll
    for (int k = 0; k < 8; ++k) mx = fmaxf(mx, acc[k]);
    mx = fmaxf(mx, __shfl_xor(mx, 16, 32));        // combine the two row halves
    if (lane < 16) orow[t * 16 + lane] = mx;
  }
}

// ------------------------------------------------------------------ launch --
extern "C" void kernel_launch(void* const* d_in, const int* in_sizes, int n_in,
                              void* d_out, int out_size, void* d_ws, size_t ws_size,
                              hipStream_t stream) {
  const float* xyz    = (const float*)d_in[0];
  const float* feat   = (const float*)d_in[1];
  const int*   offset = (const int*)d_in[2];
  const float* vel    = (const float*)d_in[3];
  const float* W      = (const float*)d_in[4];

  const int N     = in_sizes[0] / 3;
  const int B     = in_sizes[2];
  const int n     = N / B;          // 8192
  const int m     = n / STRIDE;     // 2048
  const int total = B * m;          // 4096

  int* sampled = (int*)d_ws;        // [total]
  int* knn     = sampled + total;   // [total * NS]

  float* out      = (float*)d_out;  // concatenated tuple, float32
  float* out_xyz  = out;
  float* out_feat = out + (size_t)total * 3;
  float* out_off  = out_feat + (size_t)total * C_OUT;
  float* out_vel  = out_off + B;

  fps_kernel<<<B, 1024, 0, stream>>>(xyz, feat, sampled, n, m);
  gather_kernel<<<(total + 255) / 256, 256, 0, stream>>>(
      xyz, vel, sampled, offset, out_xyz, out_vel, out_off, n, m, total, B);
  knn_kernel<<<(total + 7) / 8, 256, 0, stream>>>(xyz, sampled, knn, n, m, total);
  group_gemm_kernel<<<(total + 7) / 8, 256, 0, stream>>>(
      feat, W, knn, out_feat, n, m, total);
}
